// HIM_41394894799817
// MI455X (gfx1250) — compile-verified, matrix operand order checked by reading, not run on hardware
//
#include <hip/hip_runtime.h>

// ---------------------------------------------------------------------------
// Fused cross-attention for MI455X (gfx1250, wave32, WMMA + async LDS copies).
//   B=4, C=192, H=W=128 (N=16384), heads=6, hd=32, M=256, KVcols=384
// ---------------------------------------------------------------------------

typedef __attribute__((ext_vector_type(16))) _Float16 v16h;
typedef __attribute__((ext_vector_type(8)))  _Float16 v8h;
typedef __attribute__((ext_vector_type(8)))  float    v8f;
typedef int v4i __attribute__((vector_size(16)));

typedef __attribute__((address_space(1))) v4i* gvec4p;
typedef __attribute__((address_space(3))) v4i* lvec4p;

#define BATCH   4
#define CDIM    192
#define NHEADS  6
#define HD      32
#define MLEN    256
#define NTOT    16384
#define KVCOLS  384

#if __has_builtin(__builtin_amdgcn_global_load_async_to_lds_b128)
#define HAS_ASYNC_LDS 1
#else
#define HAS_ASYNC_LDS 0
#endif

// 16-byte global -> LDS copy: async (no VGPR round-trip) when available.
__device__ __forceinline__ void cp16(void* l, const void* g) {
#if HAS_ASYNC_LDS
    __builtin_amdgcn_global_load_async_to_lds_b128(
        (gvec4p)(void*)g, (lvec4p)l, 0, 0);
#else
    *reinterpret_cast<float4*>(l) = *reinterpret_cast<const float4*>(g);
#endif
}

__device__ __forceinline__ void async_wait0() {
#if HAS_ASYNC_LDS
#if __has_builtin(__builtin_amdgcn_s_wait_asynccnt)
    __builtin_amdgcn_s_wait_asynccnt(0);
#else
    asm volatile("s_wait_asynccnt 0x0" ::: "memory");
#endif
#endif
}

__device__ __forceinline__ v16h ldfrag(const _Float16* p0, const _Float16* p1) {
    // 16-bit WMMA A/B fragment: per lane two runs of 8 contiguous halves.
    union { v16h v; v8h h[2]; } u;
    u.h[0] = *reinterpret_cast<const v8h*>(p0);
    u.h[1] = *reinterpret_cast<const v8h*>(p1);
    return u.v;
}

__device__ __forceinline__ v8f wmma_f16(v16h a, v16h b, v8f c) {
    return __builtin_amdgcn_wmma_f32_16x16x32_f16(false, a, false, b, (short)0, c,
                                                  false, false);
}

// ===========================================================================
// Kernel 0: one-shot weight prep -> transposed f16 weights in workspace.
//   WqT[n][k], WpT[n][k] (192x192), WkvT[col][k] (384x192)
// ===========================================================================
__global__ __launch_bounds__(256)
void prep_weights(const float* __restrict__ Wq, const float* __restrict__ Wp,
                  const float* __restrict__ Wkv,
                  _Float16* __restrict__ WqT, _Float16* __restrict__ WpT,
                  _Float16* __restrict__ WkvT) {
    int idx = blockIdx.x * 256 + threadIdx.x;      // 147456 total
    if (idx < 36864) {
        int k = idx / CDIM, n = idx % CDIM;
        WqT[n * CDIM + k] = (_Float16)Wq[idx];
    } else if (idx < 73728) {
        int i = idx - 36864;
        int k = i / CDIM, n = i % CDIM;
        WpT[n * CDIM + k] = (_Float16)Wp[i];
    } else {
        int i = idx - 73728;
        int k = i / KVCOLS, col = i % KVCOLS;
        WkvT[col * CDIM + k] = (_Float16)Wkv[i];
    }
}

// ===========================================================================
// Kernel 1: kv = LN(prior) @ Wkv  ->  K [b][h][m][d] f16,  Vt [b][h][d][m] f16
//   grid = B*M/16 = 64 blocks, 128 threads (4 waves).
// ===========================================================================
struct SmemKV {
    _Float16 sA[16 * 200];                 // LN(prior) tile, row-major, pad 200
    union {
        _Float16 sWT[KVCOLS * CDIM];       // WkvT image: [col][k], stride 192
        float    pr[16 * CDIM];            // raw prior tile (phase 1 only)
    } u;
};

__global__ __launch_bounds__(128, 1)
void kv_kernel(const float* __restrict__ prior,
               const float* __restrict__ g2, const float* __restrict__ b2,
               const _Float16* __restrict__ WkvT,
               _Float16* __restrict__ Kbuf, _Float16* __restrict__ Vt) {
    __shared__ SmemKV sm;
    const int tid  = threadIdx.x;
    const int lane = tid & 31;
    const int wid  = tid >> 5;
    const int b    = blockIdx.x >> 4;
    const int m0   = (blockIdx.x & 15) << 4;

    // ---- async copy 16x192 prior tile (contiguous 12288 B) ----
    {
        const char* src = (const char*)(prior + ((size_t)(b * MLEN + m0)) * CDIM);
        char* dst = (char*)&sm.u.pr[0];
        for (int it = 0; it < 6; ++it) {
            int idx = it * 128 + tid;
            cp16(dst + idx * 16, src + idx * 16);
        }
        async_wait0();
    }
    __syncthreads();

    // ---- LayerNorm: 8 threads per row, shuffle-reduce within 8-lane groups --
    {
        const int r = tid >> 3, j = tid & 7;
        float s = 0.f, s2 = 0.f;
        for (int q = 0; q < 24; ++q) {
            float v = sm.u.pr[r * CDIM + j * 24 + q];
            s += v; s2 += v * v;
        }
        s  += __shfl_xor(s, 1);  s2 += __shfl_xor(s2, 1);
        s  += __shfl_xor(s, 2);  s2 += __shfl_xor(s2, 2);
        s  += __shfl_xor(s, 4);  s2 += __shfl_xor(s2, 4);
        float mu   = s * (1.0f / CDIM);
        float var  = s2 * (1.0f / CDIM) - mu * mu;
        float rstd = rsqrtf(var + 1e-5f);
        for (int q = 0; q < 24; ++q) {
            int c = j * 24 + q;
            float v = sm.u.pr[r * CDIM + c];
            sm.sA[r * 200 + c] = (_Float16)((v - mu) * rstd * g2[c] + b2[c]);
        }
    }
    __syncthreads();   // pr region dead

    // ---- async copy WkvT image (contiguous 147456 B) ----
    {
        const char* src = (const char*)WkvT;
        char* dst = (char*)&sm.u.sWT[0];
        for (int it = 0; it < 72; ++it) {
            int idx = it * 128 + tid;
            cp16(dst + idx * 16, src + idx * 16);
        }
        async_wait0();
    }
    __syncthreads();

    // ---- WMMA: out(16 x 384) = sA(16x192) @ Wkv(192x384) ----
    const int kb = (lane >> 4) << 3;   // 0 or 8
    v16h afr[6];
#pragma unroll
    for (int kc = 0; kc < 6; ++kc) {
        const _Float16* ap = &sm.sA[(lane & 15) * 200 + kc * 32 + kb];
        afr[kc] = ldfrag(ap, ap + 16);
    }
    for (int i = 0; i < 6; ++i) {
        const int col0 = (wid * 6 + i) * 16;
        v8f acc = {};
#pragma unroll
        for (int kc = 0; kc < 6; ++kc) {
            const _Float16* wp = &sm.u.sWT[(col0 + (lane & 15)) * CDIM + kc * 32 + kb];
            acc = wmma_f16(afr[kc], ldfrag(wp, wp + 16), acc);
        }
        const int col = col0 + (lane & 15);
#pragma unroll
        for (int r = 0; r < 8; ++r) {
            const int m = m0 + r + ((lane >> 4) << 3);
            _Float16 val = (_Float16)acc[r];
            if (col0 < CDIM) {           // K half: [b][h][m][d], d contiguous
                int h = col >> 5, d = col & 31;
                Kbuf[(((size_t)(b * NHEADS + h)) * MLEN + m) * HD + d] = val;
            } else {                     // V half -> transposed: [b][h][d][m]
                int c2 = col - CDIM;
                int h = c2 >> 5, d = c2 & 31;
                Vt[(((size_t)(b * NHEADS + h)) * HD + d) * MLEN + m] = val;
            }
        }
    }
}

// ===========================================================================
// Kernel 2: fused LN(x) -> Qproj -> 6-head attention -> out-proj -> residual
//   grid = B * N/128 = 512 blocks, 256 threads (8 waves, 16 query rows/wave).
// ===========================================================================
struct SmemAT {
    _Float16 sX[128 * 200];                // LN(x) tile, then q, then ctx
    union {
        _Float16 sW[CDIM * CDIM];          // WqT then WpT image: [col][k]
        _Float16 sP[8 * 16 * 264];         // per-wave softmax probs
        float    xt[CDIM * 128];           // raw x tile [c][n] (phase 1)
        float    ob[128 * 193];            // out f32 staging (final phase)
    } u;
    float mu[128];
    float rstd[128];
};

__global__ __launch_bounds__(256, 1)
void attn_fused(const float* __restrict__ x,
                const float* __restrict__ g1, const float* __restrict__ b1,
                const _Float16* __restrict__ WqT, const _Float16* __restrict__ WpT,
                const float* __restrict__ bp,
                const _Float16* __restrict__ Kbuf,
                const _Float16* __restrict__ Vt,
                float* __restrict__ y) {
    __shared__ SmemAT sm;
    const int tid  = threadIdx.x;
    const int lane = tid & 31;
    const int wid  = tid >> 5;
    const int b    = blockIdx.x >> 7;
    const int n0   = (blockIdx.x & 127) * 128;
    const int kb   = (lane >> 4) << 3;       // 0 / 8 : fragment K sub-run
    const int rofs = (lane >> 4) << 3;       // D-tile row offset for hi lanes
    const int row0 = wid * 16;

    // ---- phase 1: async stage x tile [c=192][n=128] f32 (512 B rows) ----
    {
        const char* xrow0 = (const char*)(x + ((size_t)(b * CDIM)) * NTOT + n0);
        char* dst = (char*)&sm.u.xt[0];
        for (int it = 0; it < 24; ++it) {
            int idx = it * 256 + tid;          // 6144 chunks of 16 B
            int c = idx >> 5, q = idx & 31;
            cp16(dst + idx * 16, xrow0 + (size_t)c * (NTOT * 4) + q * 16);
        }
        async_wait0();
    }
    __syncthreads();

    // ---- phase 2: per-query LN stats ----
    if (tid < 128) {
        float s = 0.f, s2 = 0.f;
        for (int c = 0; c < CDIM; ++c) {
            float v = sm.u.xt[c * 128 + tid];
            s += v; s2 += v * v;
        }
        float m = s * (1.0f / CDIM);
        sm.mu[tid]   = m;
        sm.rstd[tid] = rsqrtf(s2 * (1.0f / CDIM) - m * m + 1e-5f);
    }
    __syncthreads();

    // ---- phase 3: normalized f16 tile, transposed to [n][c] ----
    for (int it = 0; it < 96; ++it) {
        int idx = it * 256 + tid;
        int c = idx >> 7, n = idx & 127;
        float v = (sm.u.xt[c * 128 + n] - sm.mu[n]) * sm.rstd[n] * g1[c] + b1[c];
        sm.sX[n * 200 + c] = (_Float16)v;
    }
    __syncthreads();   // xt dead

    // ---- phase 4: async copy WqT image (73728 B contiguous) ----
    {
        const char* src = (const char*)WqT;
        char* dst = (char*)&sm.u.sW[0];
        for (int it = 0; it < 18; ++it) {
            int idx = it * 256 + tid;
            cp16(dst + idx * 16, src + idx * 16);
        }
        async_wait0();
    }
    __syncthreads();

    // ---- phase 5: q = LN(x) @ Wq, wave's 16 rows, then overwrite sX rows ----
    {
        v8f acc[12];
#pragma unroll
        for (int ct = 0; ct < 12; ++ct) acc[ct] = (v8f){};
#pragma unroll
        for (int kc = 0; kc < 6; ++kc) {
            const _Float16* ap = &sm.sX[(row0 + (lane & 15)) * 200 + kc * 32 + kb];
            v16h a = ldfrag(ap, ap + 16);
#pragma unroll
            for (int ct = 0; ct < 12; ++ct) {
                const _Float16* wp =
                    &sm.u.sW[(ct * 16 + (lane & 15)) * CDIM + kc * 32 + kb];
                acc[ct] = wmma_f16(a, ldfrag(wp, wp + 16), acc[ct]);
            }
        }
#pragma unroll
        for (int ct = 0; ct < 12; ++ct)
#pragma unroll
            for (int r = 0; r < 8; ++r)
                sm.sX[(row0 + r + rofs) * 200 + ct * 16 + (lane & 15)] =
                    (_Float16)acc[ct][r];
    }
    __syncthreads();   // WqT consumed; sP region may now be used

    // ---- phase 6: attention per head ----
    _Float16* myP = &sm.u.sP[wid * 16 * 264];
    const float scale = 0.17677669529663687f;   // 1/sqrt(32)

    for (int h = 0; h < NHEADS; ++h) {
        const _Float16* Kh = Kbuf + ((size_t)(b * NHEADS + h)) * MLEN * HD;
        const _Float16* Vh = Vt + ((size_t)(b * NHEADS + h)) * HD * MLEN;
        __builtin_prefetch(Kh, 0, 0);
        __builtin_prefetch(Vh, 0, 0);

        const _Float16* qp = &sm.sX[(row0 + (lane & 15)) * 200 + h * 32 + kb];
        v16h qf = ldfrag(qp, qp + 16);

        // scores: 16 WMMA tiles over M=256
        v8f sc[16];
#pragma unroll
        for (int mt = 0; mt < 16; ++mt) {
            const _Float16* kp = Kh + (mt * 16 + (lane & 15)) * HD + kb;
            v8f z = {};
            sc[mt] = wmma_f16(qf, ldfrag(kp, kp + 16), z);
        }
#pragma unroll
        for (int mt = 0; mt < 16; ++mt) sc[mt] = sc[mt] * scale;

        // row-wise softmax (rows live in (reg, 16-lane half) slots)
        float rmax[8], rsum[8], rinv[8];
#pragma unroll
        for (int r = 0; r < 8; ++r) rmax[r] = -3.4e38f;
#pragma unroll
        for (int mt = 0; mt < 16; ++mt)
#pragma unroll
            for (int r = 0; r < 8; ++r) rmax[r] = fmaxf(rmax[r], sc[mt][r]);
#pragma unroll
        for (int r = 0; r < 8; ++r) {
            rmax[r] = fmaxf(rmax[r], __shfl_xor(rmax[r], 1));
            rmax[r] = fmaxf(rmax[r], __shfl_xor(rmax[r], 2));
            rmax[r] = fmaxf(rmax[r], __shfl_xor(rmax[r], 4));
            rmax[r] = fmaxf(rmax[r], __shfl_xor(rmax[r], 8));
            rsum[r] = 0.f;
        }
#pragma unroll
        for (int mt = 0; mt < 16; ++mt)
#pragma unroll
            for (int r = 0; r < 8; ++r) {
                float e = __expf(sc[mt][r] - rmax[r]);
                sc[mt][r] = e;
                rsum[r] += e;
            }
#pragma unroll
        for (int r = 0; r < 8; ++r) {
            rsum[r] += __shfl_xor(rsum[r], 1);
            rsum[r] += __shfl_xor(rsum[r], 2);
            rsum[r] += __shfl_xor(rsum[r], 4);
            rsum[r] += __shfl_xor(rsum[r], 8);
            rinv[r] = 1.0f / rsum[r];
        }
        // probs -> per-wave LDS (re-fragment D-layout into A-layout)
#pragma unroll
        for (int mt = 0; mt < 16; ++mt)
#pragma unroll
            for (int r = 0; r < 8; ++r)
                myP[(r + rofs) * 264 + mt * 16 + (lane & 15)] =
                    (_Float16)(sc[mt][r] * rinv[r]);

        // out_h = P(16x256) @ V_h(256x32)
        v8f o0 = {}, o1 = {};
#pragma unroll
        for (int mc = 0; mc < 8; ++mc) {
            const _Float16* pp = &myP[(lane & 15) * 264 + mc * 32 + kb];
            v16h pf = ldfrag(pp, pp + 16);
            const _Float16* v0 = Vh + (lane & 15) * MLEN + mc * 32 + kb;
            const _Float16* v1 = Vh + (16 + (lane & 15)) * MLEN + mc * 32 + kb;
            o0 = wmma_f16(pf, ldfrag(v0, v0 + 16), o0);
            o1 = wmma_f16(pf, ldfrag(v1, v1 + 16), o1);
        }
        // ctx back into own sX rows, columns [h*32, h*32+32)
#pragma unroll
        for (int r = 0; r < 8; ++r) {
            int rr = (row0 + r + rofs) * 200 + h * 32 + (lane & 15);
            sm.sX[rr]      = (_Float16)o0[r];
            sm.sX[rr + 16] = (_Float16)o1[r];
        }
    }
    __syncthreads();   // sP dead

    // ---- phase 7: async copy WpT image ----
    {
        const char* src = (const char*)WpT;
        char* dst = (char*)&sm.u.sW[0];
        for (int it = 0; it < 18; ++it) {
            int idx = it * 256 + tid;
            cp16(dst + idx * 16, src + idx * 16);
        }
        async_wait0();
    }
    __syncthreads();

    // ---- phase 8: out = ctx @ Wp + bp ----
    v8f acc[12];
#pragma unroll
    for (int ct = 0; ct < 12; ++ct) acc[ct] = (v8f){};
#pragma unroll
    for (int kc = 0; kc < 6; ++kc) {
        const _Float16* ap = &sm.sX[(row0 + (lane & 15)) * 200 + kc * 32 + kb];
        v16h a = ldfrag(ap, ap + 16);
#pragma unroll
        for (int ct = 0; ct < 12; ++ct) {
            const _Float16* wp =
                &sm.u.sW[(ct * 16 + (lane & 15)) * CDIM + kc * 32 + kb];
            acc[ct] = wmma_f16(a, ldfrag(wp, wp + 16), acc[ct]);
        }
    }
#pragma unroll
    for (int ct = 0; ct < 12; ++ct) {
        float bv = bp[ct * 16 + (lane & 15)];
#pragma unroll
        for (int r = 0; r < 8; ++r) acc[ct][r] += bv;
    }
    __syncthreads();   // all waves done reading WpT; ob overlaps sW
#pragma unroll
    for (int ct = 0; ct < 12; ++ct)
#pragma unroll
        for (int r = 0; r < 8; ++r)
            sm.u.ob[(row0 + r + rofs) * 193 + ct * 16 + (lane & 15)] = acc[ct][r];
    __syncthreads();

    // ---- phase 9: y[b][c][n] = x[b][c][n] + out[n][c]  (coalesced along n) --
    for (int it = 0; it < 96; ++it) {
        int idx = it * 256 + tid;
        int c = idx >> 7, n = idx & 127;
        size_t gi = ((size_t)(b * CDIM + c)) * NTOT + n0 + n;
        y[gi] = x[gi] + sm.u.ob[n * 193 + c];
    }
}

// ===========================================================================
extern "C" void kernel_launch(void* const* d_in, const int* in_sizes, int n_in,
                              void* d_out, int out_size, void* d_ws, size_t ws_size,
                              hipStream_t stream) {
    const float* x     = (const float*)d_in[0];
    const float* prior = (const float*)d_in[1];
    const float* ln1_g = (const float*)d_in[2];
    const float* ln1_b = (const float*)d_in[3];
    const float* ln2_g = (const float*)d_in[4];
    const float* ln2_b = (const float*)d_in[5];
    const float* Wq    = (const float*)d_in[6];
    const float* Wkv   = (const float*)d_in[7];
    const float* Wp    = (const float*)d_in[8];
    const float* bp    = (const float*)d_in[9];
    float* y = (float*)d_out;

    const size_t kv_elems = (size_t)BATCH * NHEADS * MLEN * HD;   // 196608
    _Float16* Kbuf = (_Float16*)d_ws;
    _Float16* Vt   = Kbuf + kv_elems;
    _Float16* WqT  = Vt + kv_elems;                 // 36864 elems
    _Float16* WpT  = WqT + CDIM * CDIM;             // 36864 elems
    _Float16* WkvT = WpT + CDIM * CDIM;             // 73728 elems

    prep_weights<<<576, 256, 0, stream>>>(Wq, Wp, Wkv, WqT, WpT, WkvT);
    kv_kernel<<<BATCH * MLEN / 16, 128, 0, stream>>>(prior, ln2_g, ln2_b, WkvT,
                                                     Kbuf, Vt);
    attn_fused<<<BATCH * NTOT / 128, 256, 0, stream>>>(x, ln1_g, ln1_b, WqT, WpT,
                                                       bp, Kbuf, Vt, y);
}
